// SingleHeadSelfAttention_6253472383468
// MI455X (gfx1250) — compile-verified, matrix operand order checked by reading, not run on hardware
//
#include <hip/hip_runtime.h>

// Single-head self-attention for B=8, T=2048, E=1024 (fp32 in/out).
// All GEMMs via V_WMMA_F32_16X16X32_BF16 (wave32), fp32 accum.
// K'/V' slabs staged to LDS with GLOBAL_LOAD_ASYNC_TO_LDS_B128 (ASYNCcnt).
//   ws layout: Qp bf16 [B*T,E] | Kp bf16 | Vp bf16 | S f32 [B,T,T]  (~224 MB)

constexpr int Bn = 8;
constexpr int Tn = 2048;
constexpr int En = 1024;

typedef __attribute__((ext_vector_type(16))) __bf16 v16bf;
typedef __attribute__((ext_vector_type(8)))  float  v8f;

union FragB16 {
    v16bf v;
    unsigned u[8];
    unsigned short s[16];
    __bf16 h[16];
};

// Native f32 -> bf16 (hardware convert), returned as raw bits for stores.
__device__ __forceinline__ unsigned short bf16_bits(float f) {
    union { __bf16 b; unsigned short s; } c;
    c.b = (__bf16)f;
    return c.s;
}

__device__ __forceinline__ float neg_inf_f32() {
    union { unsigned u; float f; } c; c.u = 0xFF800000u;
    return c.f;
}

// K-offset inside a 16x32 bf16 A/B fragment: VGPR v holds K pair {k0,k0+1}.
__device__ __forceinline__ int kOff(int v, int half) {
    return ((v < 4) ? (2 * v) : (8 + 2 * v)) + 8 * half;
}

__device__ __forceinline__ v8f wmma_bf16(const FragB16& a, const FragB16& b, v8f c) {
    return __builtin_amdgcn_wmma_f32_16x16x32_bf16(
        /*neg_a=*/false, a.v, /*neg_b=*/false, b.v,
        /*c_mod=*/(short)0, c, /*reuse_a=*/false, /*reuse_b=*/false);
}

// Async 16B copy global -> LDS (per lane). lds_addr = LDS byte address.
__device__ __forceinline__ void async_ld_b128(unsigned lds_addr, const void* g) {
    asm volatile("global_load_async_to_lds_b128 %0, %1, off"
                 :
                 : "v"(lds_addr), "v"((unsigned long long)(uintptr_t)g)
                 : "memory");
}
__device__ __forceinline__ void wait_async0() {
    asm volatile("s_wait_asynccnt 0x0" ::: "memory");
}

__device__ __forceinline__ unsigned lds_addr_of(const void* p) {
    return (unsigned)(uintptr_t)p;   // LDS aperture: hardware uses addr[31:0]
}

// ---------------------------------------------------------------------------
// Kernel 1: Out[n][o] = sum_e X[n][e] * W[o][e]  (fp32 in, bf16 out)
// Wave computes a 16x64 output tile. Grid: (En/64, (Bn*Tn)/128)
// ---------------------------------------------------------------------------
__global__ __launch_bounds__(256) void proj_gemm(const float* __restrict__ X,
                                                 const float* __restrict__ W,
                                                 unsigned short* __restrict__ Out) {
    const int lane = threadIdx.x & 31;
    const int wave = threadIdx.x >> 5;
    const int m0 = (blockIdx.y * 8 + wave) * 16;
    const int n0 = blockIdx.x * 64;
    const int l15 = lane & 15;
    const int half = lane >> 4;

    v8f acc[4] = {};
    for (int kb = 0; kb < En; kb += 32) {
        FragB16 a;
#pragma unroll
        for (int v = 0; v < 8; ++v) {
            const int k = kb + kOff(v, half);
            const float2 f = *(const float2*)(X + (size_t)(m0 + l15) * En + k);
            a.h[2 * v]     = (__bf16)f.x;      // native cvt
            a.h[2 * v + 1] = (__bf16)f.y;
        }
#pragma unroll
        for (int nt = 0; nt < 4; ++nt) {
            FragB16 b;
            const int n = n0 + nt * 16 + l15;
#pragma unroll
            for (int v = 0; v < 8; ++v) {
                const int k = kb + kOff(v, half);
                const float2 f = *(const float2*)(W + (size_t)n * En + k);
                b.h[2 * v]     = (__bf16)f.x;
                b.h[2 * v + 1] = (__bf16)f.y;
            }
            acc[nt] = wmma_bf16(a, b, acc[nt]);
        }
    }
#pragma unroll
    for (int nt = 0; nt < 4; ++nt)
#pragma unroll
        for (int v = 0; v < 8; ++v) {
            const int m = m0 + v + 8 * half;
            const int n = n0 + nt * 16 + l15;
            Out[(size_t)m * En + n] = bf16_bits(acc[nt][v]);
        }
}

// ---------------------------------------------------------------------------
// Kernel 2: S[b][t][s] = scale * dot(Qp[b,t,:], Kp[b,s,:]); mask==0 -> -inf
// Block: 8 waves, each wave a 32x64 tile of S (8 accumulators).
// K' k-slab (64 keys x 32 k, 4KB) async-staged to LDS per k-step.
// Grid: (Tn/64, Tn/256, Bn)
// ---------------------------------------------------------------------------
__global__ __launch_bounds__(256) void scores_gemm(const unsigned short* __restrict__ Qp,
                                                   const unsigned short* __restrict__ Kp,
                                                   const int* __restrict__ mask,
                                                   float* __restrict__ S) {
    __shared__ unsigned short Ktile[64 * 32];   // [s][k], pitch 32 elems
    const int tid  = threadIdx.x;
    const int lane = tid & 31;
    const int wave = tid >> 5;
    const int b  = blockIdx.z;
    const int t0 = (blockIdx.y * 8 + wave) * 32;
    const int s0 = blockIdx.x * 64;
    const int l15 = lane & 15;
    const int half = lane >> 4;

    const unsigned short* Qb = Qp + (size_t)b * Tn * En;
    const unsigned short* Kb = Kp + (size_t)b * Tn * En;

    // Cooperative staging: each thread owns one 16B chunk per k-step.
    const int srow = tid >> 2;             // 0..63
    const int kgrp = (tid & 3) * 8;        // 0,8,16,24
    const unsigned ldsK = lds_addr_of(Ktile + srow * 32 + kgrp);
    const unsigned short* gK = Kb + (size_t)(s0 + srow) * En + kgrp;

    v8f acc[8] = {};
    for (int kb = 0; kb < En; kb += 32) {
        __syncthreads();                   // prior LDS reads done
        async_ld_b128(ldsK, gK + kb);
        __builtin_prefetch(gK + kb + 32, 0, 0);
        wait_async0();
        __syncthreads();                   // slab visible to all waves

        FragB16 a0, a1;
#pragma unroll
        for (int v = 0; v < 8; ++v) {
            const int k = kb + kOff(v, half);
            a0.u[v] = *(const unsigned*)(Qb + (size_t)(t0 + l15) * En + k);
            a1.u[v] = *(const unsigned*)(Qb + (size_t)(t0 + 16 + l15) * En + k);
        }
#pragma unroll
        for (int nt = 0; nt < 4; ++nt) {
            FragB16 bf;
#pragma unroll
            for (int v = 0; v < 8; ++v)
                bf.u[v] = *(const unsigned*)(Ktile + (nt * 16 + l15) * 32 + kOff(v, half));
            acc[nt]     = wmma_bf16(a0, bf, acc[nt]);
            acc[nt + 4] = wmma_bf16(a1, bf, acc[nt + 4]);
        }
    }
    const float scale = 0.03125f;  // 1/sqrt(1024)
    const float ninf = neg_inf_f32();
#pragma unroll
    for (int h = 0; h < 2; ++h)
#pragma unroll
        for (int nt = 0; nt < 4; ++nt)
#pragma unroll
            for (int v = 0; v < 8; ++v) {
                const int t = t0 + h * 16 + v + 8 * half;
                const int s = s0 + nt * 16 + l15;
                const float r = acc[h * 4 + nt][v];
                const float val = (mask[(size_t)t * Tn + s] == 0) ? ninf : r * scale;
                S[((size_t)b * Tn + t) * Tn + s] = val;
            }
}

// ---------------------------------------------------------------------------
// Kernel 3: row softmax over Tn=2048; writes bf16 P in place at row base.
// One 256-thread block per row; LDS tree reductions. Grid: Bn*Tn blocks.
// ---------------------------------------------------------------------------
__global__ __launch_bounds__(256) void softmax_rows(float* __restrict__ S) {
    __shared__ float red[256];
    const int tid = threadIdx.x;
    float* Sr = S + (size_t)blockIdx.x * Tn;
    unsigned short* Pr = (unsigned short*)Sr;

    float vals[8];
    float m = neg_inf_f32();
#pragma unroll
    for (int i = 0; i < 8; ++i) {
        vals[i] = Sr[tid + i * 256];
        m = fmaxf(m, vals[i]);
    }
    red[tid] = m;
    __syncthreads();
    for (int off = 128; off > 0; off >>= 1) {
        if (tid < off) red[tid] = fmaxf(red[tid], red[tid + off]);
        __syncthreads();
    }
    m = red[0];
    __syncthreads();

    float sum = 0.f;
#pragma unroll
    for (int i = 0; i < 8; ++i) {
        vals[i] = __expf(vals[i] - m);
        sum += vals[i];
    }
    red[tid] = sum;
    __syncthreads();
    for (int off = 128; off > 0; off >>= 1) {
        if (tid < off) red[tid] += red[tid + off];
        __syncthreads();
    }
    const float inv = 1.0f / red[0];
#pragma unroll
    for (int i = 0; i < 8; ++i)
        Pr[tid + i * 256] = bf16_bits(vals[i] * inv);
}

// ---------------------------------------------------------------------------
// Kernel 4: O[b][t][o] = sum_s P[b,t,s] * Vp[b,s,o]   (bf16 x bf16 -> f32)
// P rows are bf16 packed at the start of each S row (pitch 2*Tn bf16 elems).
// Block: 8 waves, each wave a 32x64 tile. V' slab (32 k x 64 n) async-staged.
// Grid: (En/64, Tn/256, Bn)
// ---------------------------------------------------------------------------
__global__ __launch_bounds__(256) void out_gemm(const float* __restrict__ Sws,
                                                const unsigned short* __restrict__ Vp,
                                                float* __restrict__ O) {
    __shared__ unsigned short Vtile[32 * 64];   // [k][n], pitch 64 elems
    const int tid  = threadIdx.x;
    const int lane = tid & 31;
    const int wave = tid >> 5;
    const int b  = blockIdx.z;
    const int t0 = (blockIdx.y * 8 + wave) * 32;
    const int n0 = blockIdx.x * 64;
    const int l15 = lane & 15;
    const int half = lane >> 4;

    const unsigned short* Pb = (const unsigned short*)(Sws + (size_t)b * Tn * Tn);
    const unsigned short* Vb = Vp + (size_t)b * Tn * En;

    // Cooperative staging: each thread owns one 16B chunk per k-step.
    const int krow = tid >> 3;             // 0..31
    const int ngrp = (tid & 7) * 8;        // 0..56
    const unsigned ldsV = lds_addr_of(Vtile + krow * 64 + ngrp);
    const unsigned short* gV = Vb + (size_t)krow * En + n0 + ngrp;

    v8f acc[8] = {};
    for (int kb = 0; kb < Tn; kb += 32) {
        __syncthreads();
        async_ld_b128(ldsV, gV + (size_t)kb * En);
        __builtin_prefetch(gV + (size_t)(kb + 32) * En, 0, 0);
        wait_async0();
        __syncthreads();

        FragB16 a0, a1;
#pragma unroll
        for (int v = 0; v < 8; ++v) {
            const int k = kb + kOff(v, half);
            a0.u[v] = *(const unsigned*)(Pb + (size_t)(t0 + l15) * (2 * Tn) + k);
            a1.u[v] = *(const unsigned*)(Pb + (size_t)(t0 + 16 + l15) * (2 * Tn) + k);
        }
#pragma unroll
        for (int nt = 0; nt < 4; ++nt) {
            FragB16 bf;
            const int n = nt * 16 + l15;
#pragma unroll
            for (int v = 0; v < 8; ++v) {
                const int k = kOff(v, half);
                const unsigned lo = Vtile[k * 64 + n];
                const unsigned hi = Vtile[(k + 1) * 64 + n];
                bf.u[v] = lo | (hi << 16);
            }
            acc[nt]     = wmma_bf16(a0, bf, acc[nt]);
            acc[nt + 4] = wmma_bf16(a1, bf, acc[nt + 4]);
        }
    }
#pragma unroll
    for (int h = 0; h < 2; ++h)
#pragma unroll
        for (int nt = 0; nt < 4; ++nt)
#pragma unroll
            for (int v = 0; v < 8; ++v) {
                const int t = t0 + h * 16 + v + 8 * half;
                const int n = n0 + nt * 16 + l15;
                O[((size_t)b * Tn + t) * En + n] = acc[h * 4 + nt][v];
            }
}

// ---------------------------------------------------------------------------
extern "C" void kernel_launch(void* const* d_in, const int* in_sizes, int n_in,
                              void* d_out, int out_size, void* d_ws, size_t ws_size,
                              hipStream_t stream) {
    const float* q    = (const float*)d_in[0];
    const float* k    = (const float*)d_in[1];
    const float* v    = (const float*)d_in[2];
    const int*   mask = (const int*)d_in[3];
    const float* Wq   = (const float*)d_in[4];
    const float* Wk   = (const float*)d_in[5];
    const float* Wv   = (const float*)d_in[6];
    float* out = (float*)d_out;

    unsigned short* Qp = (unsigned short*)d_ws;
    unsigned short* Kp = Qp + (size_t)Bn * Tn * En;
    unsigned short* Vp = Kp + (size_t)Bn * Tn * En;
    float*          S  = (float*)(Vp + (size_t)Bn * Tn * En);

    const dim3 blk(256);

    // QKV projections: M=Bn*Tn rows, 128 rows per block, 64 cols per block.
    const dim3 gProj(En / 64, (Bn * Tn) / 128);
    hipLaunchKernelGGL(proj_gemm, gProj, blk, 0, stream, q, Wq, Qp);
    hipLaunchKernelGGL(proj_gemm, gProj, blk, 0, stream, k, Wk, Kp);
    hipLaunchKernelGGL(proj_gemm, gProj, blk, 0, stream, v, Wv, Vp);

    // Scores + mask: 256 t-rows x 64 s-cols per block.
    const dim3 gS(Tn / 64, Tn / 256, Bn);
    hipLaunchKernelGGL(scores_gemm, gS, blk, 0, stream, Qp, Kp, mask, S);

    // Row softmax (P bf16 in place).
    hipLaunchKernelGGL(softmax_rows, dim3(Bn * Tn), blk, 0, stream, S);

    // Output GEMM: 256 t-rows x 64 o-cols per block.
    const dim3 gO(En / 64, Tn / 256, Bn);
    hipLaunchKernelGGL(out_gemm, gO, blk, 0, stream, S, Vp, out);
}
